// Attention_82240033784203
// MI455X (gfx1250) — matrix operand order repaired and checked by reading, not compiled, not run on hardware
//
#include <hip/hip_runtime.h>
#include <hip/hip_bf16.h>

// ---------------------------------------------------------------------------
// Dual-branch cross attention (prot/mol). All matmuls on the CDNA5 matrix pipe.
// bf16 dataflow for projections/FC (v_wmma_f32_16x16x32_bf16), f16 for the
// P@V stage (v_wmma_f32_16x16x32_f16), fp32 accumulation everywhere.
// B=8, S=512, D=768, H=12, DH=64.
// ---------------------------------------------------------------------------

typedef __attribute__((ext_vector_type(16))) __bf16   v16bf;
typedef __attribute__((ext_vector_type(8)))  __bf16   v8bf;
typedef __attribute__((ext_vector_type(16))) _Float16 v16h;
typedef __attribute__((ext_vector_type(8)))  _Float16 v8h;
typedef __attribute__((ext_vector_type(8)))  float    v8f;

#define NB 8
#define NS 512
#define ND 768
#define NH 12
#define NDH 64
#define MROWS (NB * NS)   // 4096

// fp32 -> bf16 round-to-nearest-even
static __device__ __forceinline__ __bf16 f2bf(float f) {
    union { float f; unsigned int u; } v;
    v.f = f;
    unsigned int r = v.u + 0x7FFFu + ((v.u >> 16) & 1u);
    unsigned short h = (unsigned short)(r >> 16);
    return __builtin_bit_cast(__bf16, h);
}

static __device__ __forceinline__ v8f wmma_bf16(v16bf a, v16bf b, v8f c) {
    return __builtin_amdgcn_wmma_f32_16x16x32_bf16(
        false, a, false, b, (short)0, c, false, false);
}
static __device__ __forceinline__ v8f wmma_f16(v16h a, v16h b, v8f c) {
    return __builtin_amdgcn_wmma_f32_16x16x32_f16(
        false, a, false, b, (short)0, c, false, false);
}

static __device__ __forceinline__ v16bf cat8b(v8bf lo, v8bf hi) {
    return __builtin_shufflevector(lo, hi, 0, 1, 2, 3, 4, 5, 6, 7,
                                   8, 9, 10, 11, 12, 13, 14, 15);
}
static __device__ __forceinline__ v16h cat8h(v8h lo, v8h hi) {
    return __builtin_shufflevector(lo, hi, 0, 1, 2, 3, 4, 5, 6, 7,
                                   8, 9, 10, 11, 12, 13, 14, 15);
}
static __device__ __forceinline__ v8bf ld8b(const __bf16* p) {
    return *reinterpret_cast<const v8bf*>(p);
}
static __device__ __forceinline__ v8h ld8h(const _Float16* p) {
    return *reinterpret_cast<const v8h*>(p);
}

// ---------------------------------------------------------------------------
// Elementwise fp32 -> bf16 (4 elems / thread)
// ---------------------------------------------------------------------------
__global__ void __launch_bounds__(256)
cvt_bf16_kernel(const float* __restrict__ in, __bf16* __restrict__ out, int n) {
    int i = (blockIdx.x * 256 + threadIdx.x) * 4;
    if (i + 3 < n) {
        float4 f = *reinterpret_cast<const float4*>(in + i);
        out[i + 0] = f2bf(f.x);
        out[i + 1] = f2bf(f.y);
        out[i + 2] = f2bf(f.z);
        out[i + 3] = f2bf(f.w);
    } else {
        for (int j = i; j < n; ++j) out[j] = f2bf(in[j]);
    }
}

// ---------------------------------------------------------------------------
// Weight transpose + convert:  Wt[n*K + k] = bf16(W[k*N + n])
// ---------------------------------------------------------------------------
__global__ void __launch_bounds__(256)
transpose_cvt_kernel(const float* __restrict__ W, __bf16* __restrict__ Wt,
                     int K, int N) {
    int idx = blockIdx.x * 256 + threadIdx.x;
    if (idx >= K * N) return;
    int n = idx / K;
    int k = idx - n * K;
    Wt[idx] = f2bf(W[(size_t)k * N + n]);
}

// ---------------------------------------------------------------------------
// GEMM: C[M,N] = A[M,K] @ W[K,N] + bias[N], weights pre-transposed (Wt[N,K]).
// A bf16 row-major, fp32 accumulation. One wave computes a 32x32 tile.
// Software-pipelined k-loop: next k-step's 8x16B loads issue before the
// current step's 4 WMMAs so VMEM latency hides under the matrix pipe.
// OUT_MODE 0: fp32 row-major (ldc)      — final outputs
// OUT_MODE 1: bf16 row-major (ldc)      — intermediates
// OUT_MODE 2: f16 head-transposed V:    Vt[(b*ND + n)*NS + s], 16B stores
// ---------------------------------------------------------------------------
template <int OUT_MODE>
__global__ void __launch_bounds__(32)
gemm_bf16_kernel(const __bf16* __restrict__ A, int lda,
                 const __bf16* __restrict__ Wt, int ldwt,
                 const float* __restrict__ bias,
                 void* __restrict__ Cout, int ldc, int K)
{
    const int n0 = blockIdx.x * 32;
    const int m0 = blockIdx.y * 32;
    const int lane = threadIdx.x;

    const int rowA = lane & 15;
    const int kbA  = (lane >> 4) * 8;    // A frag: K-halves per lane group
    const int colB = lane & 15;
    const int kbB  = (lane >> 4) * 8;    // B frag: K-halves per lane group
    const int rbase = (lane >> 4) * 8;   // C/D: lanes 16-31 hold M=8..15

    v8f acc00 = {}, acc01 = {}, acc10 = {}, acc11 = {};

    const __bf16* pa0 = A + (size_t)(m0 + rowA) * lda + kbA;
    const __bf16* pa1 = A + (size_t)(m0 + 16 + rowA) * lda + kbA;
    const __bf16* pb0 = Wt + (size_t)(n0 + colB) * ldwt + kbB;
    const __bf16* pb1 = Wt + (size_t)(n0 + 16 + colB) * ldwt + kbB;

    // prologue: fragments for k0 = 0
    v16bf a0 = cat8b(ld8b(pa0), ld8b(pa0 + 16));
    v16bf a1 = cat8b(ld8b(pa1), ld8b(pa1 + 16));
    v16bf b0 = cat8b(ld8b(pb0), ld8b(pb0 + 16));
    v16bf b1 = cat8b(ld8b(pb1), ld8b(pb1 + 16));

    for (int k0 = 0; k0 < K; k0 += 32) {
        v16bf na0, na1, nb0, nb1;
        const int kn = k0 + 32;
        if (kn < K) {   // issue next step's loads before this step's WMMAs
            na0 = cat8b(ld8b(pa0 + kn), ld8b(pa0 + kn + 16));
            na1 = cat8b(ld8b(pa1 + kn), ld8b(pa1 + kn + 16));
            nb0 = cat8b(ld8b(pb0 + kn), ld8b(pb0 + kn + 16));
            nb1 = cat8b(ld8b(pb1 + kn), ld8b(pb1 + kn + 16));
        }
        acc00 = wmma_bf16(a0, b0, acc00);
        acc01 = wmma_bf16(a0, b1, acc01);
        acc10 = wmma_bf16(a1, b0, acc10);
        acc11 = wmma_bf16(a1, b1, acc11);
        a0 = na0; a1 = na1; b0 = nb0; b1 = nb1;
    }

    const float bv0 = bias[n0 + colB];
    const float bv1 = bias[n0 + 16 + colB];

#pragma unroll
    for (int mt = 0; mt < 2; ++mt) {
#pragma unroll
        for (int nt = 0; nt < 2; ++nt) {
            v8f acc = (mt == 0) ? (nt == 0 ? acc00 : acc01)
                                : (nt == 0 ? acc10 : acc11);
            const float bv = nt ? bv1 : bv0;
            const int n = n0 + nt * 16 + colB;
            const int mbase = m0 + mt * 16 + rbase;

            if (OUT_MODE == 0) {
                float* C = (float*)Cout;
#pragma unroll
                for (int r = 0; r < 8; ++r)
                    C[(size_t)(mbase + r) * ldc + n] = acc[r] + bv;
            } else if (OUT_MODE == 1) {
                __bf16* C = (__bf16*)Cout;
#pragma unroll
                for (int r = 0; r < 8; ++r)
                    C[(size_t)(mbase + r) * ldc + n] = f2bf(acc[r] + bv);
            } else {
                // Vt[(b*ND + n)*NS + s], 8 consecutive s per lane -> 16B store
                _Float16* C = (_Float16*)Cout;
                const int b = mbase >> 9;        // mbase / NS
                const int s = mbase & (NS - 1);  // mbase % NS
                v8h pk;
#pragma unroll
                for (int r = 0; r < 8; ++r) pk[r] = (_Float16)(acc[r] + bv);
                *reinterpret_cast<v8h*>(C + ((size_t)b * ND + n) * NS + s) = pk;
            }
        }
    }
}

// ---------------------------------------------------------------------------
// Attention: one wave per (16-row q-tile, head, batch).
// Q,K bf16 [B,S,D] row-major; V f16 head-transposed Vt[(b*ND + n)*NS + s].
// Single 16KB f16 score strip in LDS (high occupancy), softmax in place
// (2 lanes/row, 8-wide vectorized, shfl combine). Unnormalized f16 probs feed
// f16 WMMA; 1/rowsum applied in fp32 at the epilogue. Output bf16 -> concat.
// ---------------------------------------------------------------------------
__global__ void __launch_bounds__(32)
attention_kernel(const __bf16* __restrict__ Q,
                 const __bf16* __restrict__ Km,
                 const _Float16* __restrict__ Vt,
                 __bf16* __restrict__ Out,
                 int out_col_off, int ldc)
{
    __shared__ __align__(16) _Float16 sc[16 * NS];   // 16 KB scores -> probs
    __shared__ float sinv[16];

    const int qt   = blockIdx.x;
    const int h    = blockIdx.y;
    const int bb   = blockIdx.z;
    const int lane = threadIdx.x;

    const __bf16*   qp = Q  + (size_t)bb * NS * ND + (size_t)h * NDH;
    const __bf16*   kp = Km + (size_t)bb * NS * ND + (size_t)h * NDH;
    const _Float16* vp = Vt + ((size_t)bb * ND + (size_t)h * NDH) * NS;

    const int rowA = lane & 15;
    const int kbA  = (lane >> 4) * 8;
    const int colB = lane & 15;
    const int kbB  = (lane >> 4) * 8;
    const int rbase = (lane >> 4) * 8;

    // Q tile fragments (dims 0..31, 32..63): contiguous bf16 vector loads
    const __bf16* qrow = qp + (size_t)(qt * 16 + rowA) * ND;
    v16bf aq0 = cat8b(ld8b(qrow + kbA),      ld8b(qrow + 16 + kbA));
    v16bf aq1 = cat8b(ld8b(qrow + 32 + kbA), ld8b(qrow + 48 + kbA));

    // scores = (Q @ K^T) / sqrt(64), stored f16
    for (int j = 0; j < NS / 16; ++j) {
        const __bf16* krow = kp + (size_t)(j * 16 + colB) * ND;
        __builtin_prefetch(krow + (size_t)16 * ND, 0, 3);  // next K tile, near
        v16bf bk0 = cat8b(ld8b(krow + kbB),      ld8b(krow + kbB + 16));
        v16bf bk1 = cat8b(ld8b(krow + 32 + kbB), ld8b(krow + 32 + kbB + 16));
        v8f c = {};
        c = wmma_bf16(aq0, bk0, c);
        c = wmma_bf16(aq1, bk1, c);
#pragma unroll
        for (int r = 0; r < 8; ++r)
            sc[(rbase + r) * NS + j * 16 + colB] = (_Float16)(c[r] * 0.125f);
    }
    __syncthreads();

    // softmax in place: 2 lanes per row, 8-wide vectorized
    {
        const int r    = lane & 15;
        const int half = lane >> 4;
        _Float16* row = sc + r * NS + half * (NS / 2);

        float m = -3.4e38f;
        for (int i = 0; i < NS / 2; i += 8) {
            v8h x = ld8h(row + i);
#pragma unroll
            for (int t = 0; t < 8; ++t) m = fmaxf(m, (float)x[t]);
        }
        m = fmaxf(m, __shfl_xor(m, 16));

        float s = 0.0f;
        for (int i = 0; i < NS / 2; i += 8) {
            v8h x = ld8h(row + i);
            v8h e;
#pragma unroll
            for (int t = 0; t < 8; ++t) {
                float ef = __expf((float)x[t] - m);
                s += ef;
                e[t] = (_Float16)ef;       // unnormalized prob in [0,1]
            }
            *reinterpret_cast<v8h*>(row + i) = e;
        }
        s += __shfl_xor(s, 16);
        if (half == 0) sinv[r] = 1.0f / s;
    }
    __syncthreads();

    // O = P @ V  (16x512 @ 512x64) via f16 WMMA, V contiguous via Vt layout
    v8f co0 = {}, co1 = {}, co2 = {}, co3 = {};
    const _Float16* prow = sc + rowA * NS;
    for (int kc = 0; kc < NS; kc += 32) {
        v16h ap = cat8h(ld8h(prow + kc + kbA), ld8h(prow + kc + 16 + kbA));
        const _Float16* v0 = vp + (size_t)(colB)      * NS + kc + kbB;
        const _Float16* v1 = vp + (size_t)(16 + colB) * NS + kc + kbB;
        const _Float16* v2 = vp + (size_t)(32 + colB) * NS + kc + kbB;
        const _Float16* v3 = vp + (size_t)(48 + colB) * NS + kc + kbB;
        co0 = wmma_f16(ap, cat8h(ld8h(v0), ld8h(v0 + 16)), co0);
        co1 = wmma_f16(ap, cat8h(ld8h(v1), ld8h(v1 + 16)), co1);
        co2 = wmma_f16(ap, cat8h(ld8h(v2), ld8h(v2 + 16)), co2);
        co3 = wmma_f16(ap, cat8h(ld8h(v3), ld8h(v3 + 16)), co3);
    }

    // normalize rows in fp32, store bf16 merged-head output into concat buffer
    __bf16* orow = Out + (size_t)out_col_off + (size_t)h * NDH + colB;
#pragma unroll
    for (int r = 0; r < 8; ++r) {
        const float inv = sinv[rbase + r];
        __bf16* o = orow + (size_t)((size_t)bb * NS + qt * 16 + rbase + r) * ldc;
        o[0]  = f2bf(co0[r] * inv);
        o[16] = f2bf(co1[r] * inv);
        o[32] = f2bf(co2[r] * inv);
        o[48] = f2bf(co3[r] * inv);
    }
}

// ---------------------------------------------------------------------------
// Host-side launch
// ---------------------------------------------------------------------------
extern "C" void kernel_launch(void* const* d_in, const int* in_sizes, int n_in,
                              void* d_out, int out_size, void* d_ws, size_t ws_size,
                              hipStream_t stream) {
    (void)in_sizes; (void)n_in; (void)out_size; (void)ws_size;

    const float* hs   = (const float*)d_in[0];
    const float* mol  = (const float*)d_in[1];
    const float* Wq   = (const float*)d_in[2];
    const float* bq   = (const float*)d_in[3];
    const float* Wk   = (const float*)d_in[4];
    const float* bk   = (const float*)d_in[5];
    const float* Wv   = (const float*)d_in[6];
    const float* bv   = (const float*)d_in[7];
    const float* Wqm  = (const float*)d_in[8];
    const float* bqm  = (const float*)d_in[9];
    const float* Wkm  = (const float*)d_in[10];
    const float* bkm  = (const float*)d_in[11];
    const float* Wvm  = (const float*)d_in[12];
    const float* bvm  = (const float*)d_in[13];
    const float* Wfc  = (const float*)d_in[14];
    const float* bfc  = (const float*)d_in[15];
    const float* Wfcm = (const float*)d_in[16];
    const float* bfcm = (const float*)d_in[17];
    const float* Wo   = (const float*)d_in[18];
    const float* bo   = (const float*)d_in[19];
    const float* Wom  = (const float*)d_in[20];
    const float* bom  = (const float*)d_in[21];

    // ---- workspace carve-up (bytes, 256-aligned chunks) ----
    char* wp = (char*)d_ws;
    auto alloc = [&wp](size_t bytes) {
        char* p = wp;
        wp += (bytes + 255) & ~(size_t)255;
        return p;
    };
    const size_t P = (size_t)MROWS * ND;      // 4096*768 elements

    __bf16*   hs_bf  = (__bf16*)alloc(P * 2);
    __bf16*   mol_bf = (__bf16*)alloc(P * 2);
    __bf16*   q_bf   = (__bf16*)alloc(P * 2);
    __bf16*   k_bf   = (__bf16*)alloc(P * 2);
    _Float16* vt_h   = (_Float16*)alloc(P * 2);  // head-transposed V (f16)
    __bf16*   qm_bf  = (__bf16*)alloc(P * 2);
    __bf16*   km_bf  = (__bf16*)alloc(P * 2);
    _Float16* vmt_h  = (_Float16*)alloc(P * 2);  // head-transposed Vm (f16)
    __bf16*   cat_p  = (__bf16*)alloc(2 * P * 2);
    __bf16*   cat_m  = (__bf16*)alloc(2 * P * 2);
    __bf16*   h_p    = (__bf16*)alloc(P * 2);
    __bf16*   h_m    = (__bf16*)alloc(P * 2);
    __bf16*   Wq_t   = (__bf16*)alloc((size_t)ND * ND * 2);
    __bf16*   Wk_t   = (__bf16*)alloc((size_t)ND * ND * 2);
    __bf16*   Wv_t   = (__bf16*)alloc((size_t)ND * ND * 2);
    __bf16*   Wqm_t  = (__bf16*)alloc((size_t)ND * ND * 2);
    __bf16*   Wkm_t  = (__bf16*)alloc((size_t)ND * ND * 2);
    __bf16*   Wvm_t  = (__bf16*)alloc((size_t)ND * ND * 2);
    __bf16*   Wfc_t  = (__bf16*)alloc((size_t)2 * ND * ND * 2);
    __bf16*   Wfcm_t = (__bf16*)alloc((size_t)2 * ND * ND * 2);
    __bf16*   Wo_t   = (__bf16*)alloc((size_t)ND * ND * 2);
    __bf16*   Wom_t  = (__bf16*)alloc((size_t)ND * ND * 2);

    float* out_prot = (float*)d_out;
    float* out_mol  = out_prot + P;

    // ---- 0) conversions ----
    {
        int n = (int)P;
        int g = (n / 4 + 255) / 256;
        cvt_bf16_kernel<<<g, 256, 0, stream>>>(hs, hs_bf, n);
        cvt_bf16_kernel<<<g, 256, 0, stream>>>(mol, mol_bf, n);
    }
    {
        int nKN = ND * ND, g = (nKN + 255) / 256;
        transpose_cvt_kernel<<<g, 256, 0, stream>>>(Wq,  Wq_t,  ND, ND);
        transpose_cvt_kernel<<<g, 256, 0, stream>>>(Wk,  Wk_t,  ND, ND);
        transpose_cvt_kernel<<<g, 256, 0, stream>>>(Wv,  Wv_t,  ND, ND);
        transpose_cvt_kernel<<<g, 256, 0, stream>>>(Wqm, Wqm_t, ND, ND);
        transpose_cvt_kernel<<<g, 256, 0, stream>>>(Wkm, Wkm_t, ND, ND);
        transpose_cvt_kernel<<<g, 256, 0, stream>>>(Wvm, Wvm_t, ND, ND);
        transpose_cvt_kernel<<<g, 256, 0, stream>>>(Wo,  Wo_t,  ND, ND);
        transpose_cvt_kernel<<<g, 256, 0, stream>>>(Wom, Wom_t, ND, ND);
        int nKN2 = 2 * ND * ND, g2 = (nKN2 + 255) / 256;
        transpose_cvt_kernel<<<g2, 256, 0, stream>>>(Wfc,  Wfc_t,  2 * ND, ND);
        transpose_cvt_kernel<<<g2, 256, 0, stream>>>(Wfcm, Wfcm_t, 2 * ND, ND);
    }

    dim3 blk(32, 1, 1);
    dim3 gP(ND / 32, MROWS / 32, 1);   // 24 x 128 waves of 32x32 tiles

    // ---- 1) six QKV projections (bf16 out; V head-transposed f16) ----
    gemm_bf16_kernel<1><<<gP, blk, 0, stream>>>(hs_bf,  ND, Wq_t,  ND, bq,  q_bf,  ND, ND);
    gemm_bf16_kernel<1><<<gP, blk, 0, stream>>>(hs_bf,  ND, Wk_t,  ND, bk,  k_bf,  ND, ND);
    gemm_bf16_kernel<2><<<gP, blk, 0, stream>>>(hs_bf,  ND, Wv_t,  ND, bv,  vt_h,  0,  ND);
    gemm_bf16_kernel<1><<<gP, blk, 0, stream>>>(mol_bf, ND, Wqm_t, ND, bqm, qm_bf, ND, ND);
    gemm_bf16_kernel<1><<<gP, blk, 0, stream>>>(mol_bf, ND, Wkm_t, ND, bkm, km_bf, ND, ND);
    gemm_bf16_kernel<2><<<gP, blk, 0, stream>>>(mol_bf, ND, Wvm_t, ND, bvm, vmt_h, 0,  ND);

    // ---- 2) four attentions into concat buffers (bf16) ----
    dim3 gA(NS / 16, NH, NB);
    attention_kernel<<<gA, blk, 0, stream>>>(q_bf,  k_bf,  vt_h,  cat_p, 0,  2 * ND); // a_pp
    attention_kernel<<<gA, blk, 0, stream>>>(qm_bf, k_bf,  vt_h,  cat_p, ND, 2 * ND); // a_mp
    attention_kernel<<<gA, blk, 0, stream>>>(qm_bf, km_bf, vmt_h, cat_m, 0,  2 * ND); // a_mm
    attention_kernel<<<gA, blk, 0, stream>>>(q_bf,  km_bf, vmt_h, cat_m, ND, 2 * ND); // a_pm

    // ---- 3) FC over concat (K = 1536, bf16 out) ----
    gemm_bf16_kernel<1><<<gP, blk, 0, stream>>>(cat_p, 2 * ND, Wfc_t,  2 * ND, bfc,  h_p, ND, 2 * ND);
    gemm_bf16_kernel<1><<<gP, blk, 0, stream>>>(cat_m, 2 * ND, Wfcm_t, 2 * ND, bfcm, h_m, ND, 2 * ND);

    // ---- 4) output projections (fp32 out) ----
    gemm_bf16_kernel<0><<<gP, blk, 0, stream>>>(h_p, ND, Wo_t,  ND, bo,  out_prot, ND, ND);
    gemm_bf16_kernel<0><<<gP, blk, 0, stream>>>(h_m, ND, Wom_t, ND, bom, out_mol,  ND, ND);
}
